// StateSpaceModel_66889820668434
// MI455X (gfx1250) — matrix operand order repaired and checked by reading, not compiled, not run on hardware
//
#include <hip/hip_runtime.h>
#include <hip/hip_bf16.h>

typedef __attribute__((ext_vector_type(16))) __bf16 bf16x16;
typedef __attribute__((ext_vector_type(8)))  float  f32x8;
typedef __attribute__((ext_vector_type(2)))  float  f32x2;
typedef __attribute__((ext_vector_type(4)))  unsigned int u32x4;
typedef __attribute__((ext_vector_type(8)))  int    i32x8;
typedef __attribute__((ext_vector_type(4)))  int    i32x4;

#define DEV __device__ __forceinline__

constexpr int kTT = 256;   // T
constexpr int kBB = 512;   // B
constexpr int kAD = 32;    // A_DIM
constexpr int kH  = 128;   // HID
constexpr int kG  = 512;   // 4*HID

DEV f32x8 wmma_bf16(bf16x16 a, bf16x16 b, f32x8 c) {
  return __builtin_amdgcn_wmma_f32_16x16x32_bf16(false, a, false, b, (short)0, c,
                                                 false, false);
}
DEV f32x8 wmma_f32k4(f32x2 a, f32x2 b, f32x8 c) {
  return __builtin_amdgcn_wmma_f32_16x16x4_f32(false, a, false, b, (short)0, c,
                                               false, false);
}
DEV float sigm_(float x) { return 1.0f / (1.0f + expf(-x)); }

// A-operand tile gather: lane (M=mrow, half) needs rows[k] for
//   k = kbase + 8*half + [0..8)  and  k = kbase + 16 + 8*half + [0..8)
// -> 4x float4 loads + packed bf16 converts (no scalar gather).
DEV bf16x16 gather_a_tile(const float* __restrict__ row, int kbase, int half) {
  const float* p = row + kbase + 8 * half;
  float4 u0 = *(const float4*)(p);
  float4 u1 = *(const float4*)(p + 4);
  float4 v0 = *(const float4*)(p + 16);
  float4 v1 = *(const float4*)(p + 20);
  bf16x16 a;
  a[0] = (__bf16)u0.x;  a[1] = (__bf16)u0.y;  a[2] = (__bf16)u0.z;  a[3] = (__bf16)u0.w;
  a[4] = (__bf16)u1.x;  a[5] = (__bf16)u1.y;  a[6] = (__bf16)u1.z;  a[7] = (__bf16)u1.w;
  a[8] = (__bf16)v0.x;  a[9] = (__bf16)v0.y;  a[10] = (__bf16)v0.z; a[11] = (__bf16)v0.w;
  a[12] = (__bf16)v1.x; a[13] = (__bf16)v1.y; a[14] = (__bf16)v1.z; a[15] = (__bf16)v1.w;
  return a;
}

// ---------------------------------------------------------------------------
// Tensor Data Mover: per-wave DMA of one contiguous row of `ndw` dwords from
// global memory into LDS.  D# per CDNA5 ISA ch.8 (count=1, type=2=image,
// data_size=4B, 1-D tile).  Guarded so non-TDM toolchains fall back to
// per-lane vector loads.
// ---------------------------------------------------------------------------
#if __has_builtin(__builtin_amdgcn_tensor_load_to_lds) && \
    __has_builtin(__builtin_amdgcn_s_wait_tensorcnt)
#define USE_TDM 1
#pragma message("CDNA5: TDM tensor_load_to_lds path ENABLED")
#else
#pragma message("CDNA5: TDM builtins not found, using per-lane fallback")
#endif

DEV void load_row_to_lds(float* lds_dst, const float* gsrc, unsigned ndw, int lane) {
#ifdef USE_TDM
  unsigned lds_addr = (unsigned)(unsigned long long)(uintptr_t)lds_dst;  // addr[31:0] = LDS offset
  unsigned long long ga = (unsigned long long)(uintptr_t)gsrc;
  u32x4 g0;
  g0[0] = 1u;                                   // count=1 (valid), user mode
  g0[1] = lds_addr;                             // lds_addr (bytes)
  g0[2] = (unsigned)(ga & 0xFFFFFFFFu);         // global_addr[31:0]
  g0[3] = (unsigned)((ga >> 32) & 0x1FFFFFFu)   // global_addr[56:32]
          | (2u << 30);                         // type=2 ("image")
  i32x8 g1;
  g1[0] = (int)(2u << 16);                      // data_size=2 -> 4 bytes
  g1[1] = (int)(ndw << 16);                     // tensor_dim0[15:0]
  g1[2] = (int)(1u << 16);                      // tensor_dim0 hi | tensor_dim1=1
  g1[3] = (int)(ndw << 16);                     // tile_dim0
  g1[4] = 1;                                    // tile_dim1=1
  g1[5] = (int)ndw;                             // tensor_dim0_stride
  g1[6] = 0;
  g1[7] = 0;
  i32x4 z4 = {0, 0, 0, 0};
#if __has_include(<hip/amd_detail/amd_gfx1250_TDM.h>)
  i32x8 z8 = {0, 0, 0, 0, 0, 0, 0, 0};          // therock 6-arg form
  __builtin_amdgcn_tensor_load_to_lds(g0, g1, z4, z4, z8, 0);
#else
  __builtin_amdgcn_tensor_load_to_lds(g0, g1, z4, z4, 0);  // ROCm 7.2 5-arg form
#endif
#else
  for (unsigned i = lane; i < ndw; i += 32) lds_dst[i] = gsrc[i];
#endif
}

DEV void wait_rows() {
#ifdef USE_TDM
  __builtin_amdgcn_s_wait_tensorcnt(0);
#endif
}

// ---------------------------------------------------------------------------
// Kernel 1: pack weights into bf16 WMMA B-operand layout, biases, Q, R
// Packed layout: [kt][nt][lane][e], value = W[nt*16+(lane&15)][kt*32+8*(lane>>4)+(e&7)+16*(e>>3)]
// ---------------------------------------------------------------------------
DEV void pack_bf16(const float* __restrict__ W, __bf16* __restrict__ P, int Kdim,
                   int idx) {
  int e = idx & 15, lane = (idx >> 4) & 31, nt = (idx >> 9) & 31, kt = idx >> 14;
  int n = nt * 16 + (lane & 15);
  int k = kt * 32 + 8 * (lane >> 4) + (e & 7) + 16 * (e >> 3);
  P[idx] = (__bf16)W[n * Kdim + k];
}

__global__ void prep_kernel(const float* Wih0, const float* Whh0, const float* bih0,
                            const float* bhh0, const float* Wih1, const float* Whh1,
                            const float* bih1, const float* bhh1, const float* Q_L,
                            const float* R_L, __bf16* wih0p, __bf16* whh0p,
                            __bf16* wih1p, __bf16* whh1p, float* bias0, float* bias1,
                            float* Qm, float* Rm) {
  int gid = blockIdx.x * 256 + threadIdx.x;
  if (gid < 16384) {
    pack_bf16(Wih0, wih0p, 32, gid);
  } else if (gid < 16384 + 65536) {
    pack_bf16(Whh0, whh0p, 128, gid - 16384);
  } else if (gid < 16384 + 2 * 65536) {
    pack_bf16(Wih1, wih1p, 128, gid - 16384 - 65536);
  } else if (gid < 16384 + 3 * 65536) {
    pack_bf16(Whh1, whh1p, 128, gid - 16384 - 2 * 65536);
  } else {
    int r = gid - (16384 + 3 * 65536);
    if (r < 512) {
      bias0[r] = bih0[r] + bhh0[r];
    } else if (r < 1024) {
      bias1[r - 512] = bih1[r - 512] + bhh1[r - 512];
    } else if (r < 1024 + 256) {
      int q = r - 1024, i = q >> 4, j = q & 15;
      float s = 0.f;
      for (int p = 0; p < 16; ++p) s += Q_L[i * 16 + p] * Q_L[j * 16 + p];
      Qm[q] = s + ((i == j) ? 1e-3f : 0.f);
    } else if (r < 1024 + 256 + 1024) {
      int q = r - 1280, i = q >> 5, j = q & 31;
      float s = 0.f;
      for (int p = 0; p < 32; ++p) s += R_L[i * 32 + p] * R_L[j * 32 + p];
      Rm[q] = s + ((i == j) ? 1e-3f : 0.f);
    }
  }
}

// ---------------------------------------------------------------------------
// Kernel 2: t=0 mixture matrices (prepended weights == 1 -> sum over K)
// ---------------------------------------------------------------------------
__global__ void fill_t0_kernel(const float* A_K, const float* C_K, float* outAs,
                               float* outCs) {
  int gid = blockIdx.x * 256 + threadIdx.x;
  if (gid < kBB * 256) {
    int ij = gid & 255;
    outAs[gid] = A_K[ij] + A_K[256 + ij] + A_K[512 + ij];
  } else if (gid < kBB * 256 + kBB * 512) {
    int g = gid - kBB * 256;
    int ij = g & 511;
    outCs[g] = C_K[ij] + C_K[512 + ij] + C_K[1024 + ij];
  }
}

// ---------------------------------------------------------------------------
// Kernel 3: fused 2-layer LSTM + softmax + mixture-matrix emission.
// One workgroup = 16 batch rows; 8 waves each own 64 gate columns (4 N-tiles).
// N-tile loops are NOT unrolled (unrolling spills B tiles to scratch).
// Two alternating accumulators per tile break the WMMA->WMMA D->C chain so
// the compiler does not need hazard v_nops between back-to-back WMMAs.
// ---------------------------------------------------------------------------
__global__ __launch_bounds__(256) void lstm_fused_kernel(
    const float* __restrict__ xin, const __bf16* __restrict__ wih0p,
    const __bf16* __restrict__ whh0p, const __bf16* __restrict__ wih1p,
    const __bf16* __restrict__ whh1p, const float* __restrict__ bias0,
    const float* __restrict__ bias1, const float* __restrict__ Wlin,
    const float* __restrict__ blin, const float* __restrict__ A_K,
    const float* __restrict__ C_K, float* __restrict__ outAs,
    float* __restrict__ outCs) {
  __shared__ float sh_gact[16][kG];   // 32 KB gate-activation exchange
  __shared__ float sh_h0[16][kH];     // 8 KB
  __shared__ float sh_h1[16][kH];     // 8 KB
  __shared__ float sh_b0[kG];
  __shared__ float sh_b1[kG];
  __shared__ float sh_w[16][4];

  const int tid = threadIdx.x;
  const int wv = tid >> 5, lane = tid & 31;
  const int mrow = lane & 15, half = lane >> 4;
  const int bt = blockIdx.x;
  const int rr = tid >> 4;            // row owned for c/h update
  const int jb = (tid & 15) * 8;      // hidden cols owned

  for (int i = tid; i < kG; i += 256) { sh_b0[i] = bias0[i]; sh_b1[i] = bias1[i]; }
  for (int i = tid; i < 16 * kH; i += 256) {
    (&sh_h0[0][0])[i] = 0.f;
    (&sh_h1[0][0])[i] = 0.f;
  }
  float c0[8], c1[8];
#pragma unroll
  for (int s = 0; s < 8; ++s) { c0[s] = 0.f; c1[s] = 0.f; }
  __syncthreads();

  const bool gate_tanh = (wv == 4 || wv == 5);  // wave's 64 cols lie in one gate

  for (int t = 0; t < kTT; ++t) {
    // ---------------- layer 0 ----------------
    const float* xr = xin + ((size_t)t * kBB + bt * 16 + mrow) * kAD;
    __builtin_prefetch(xr + (size_t)kBB * kAD, 0, 1);  // next step's row
    bf16x16 ax = gather_a_tile(xr, 0, half);
    bf16x16 ah[4];
#pragma unroll
    for (int kt = 0; kt < 4; ++kt) ah[kt] = gather_a_tile(&sh_h0[mrow][0], kt * 32, half);

#pragma unroll 1
    for (int i = 0; i < 4; ++i) {
      int nt = wv * 4 + i, n = nt * 16 + mrow;
      f32x8 acc0, acc1;
      float bv = sh_b0[n];
#pragma unroll
      for (int r = 0; r < 8; ++r) { acc0[r] = bv; acc1[r] = 0.f; }
      {
        const bf16x16* bp = (const bf16x16*)(wih0p + ((size_t)(nt * 32 + lane)) * 16);
        acc0 = wmma_bf16(ax, *bp, acc0);
      }
#pragma unroll
      for (int kt = 0; kt < 4; ++kt) {
        const bf16x16* bp =
            (const bf16x16*)(whh0p + (((size_t)kt * 32 + nt) * 32 + lane) * 16);
        if (kt & 1) acc0 = wmma_bf16(ah[kt], *bp, acc0);
        else        acc1 = wmma_bf16(ah[kt], *bp, acc1);
      }
#pragma unroll
      for (int r = 0; r < 8; ++r) {
        float v = acc0[r] + acc1[r];
        v = gate_tanh ? tanhf(v) : sigm_(v);
        sh_gact[r + 8 * half][n] = v;
      }
    }
    __syncthreads();
#pragma unroll
    for (int s = 0; s < 8; ++s) {
      int j = jb + s;
      float ig = sh_gact[rr][j], fg = sh_gact[rr][kH + j];
      float gg = sh_gact[rr][2 * kH + j], og = sh_gact[rr][3 * kH + j];
      float c = fg * c0[s] + ig * gg;
      c0[s] = c;
      sh_h0[rr][j] = og * tanhf(c);
    }
    __syncthreads();

    // ---------------- layer 1 ----------------
    bf16x16 a0[4], a1[4];
#pragma unroll
    for (int kt = 0; kt < 4; ++kt) {
      a0[kt] = gather_a_tile(&sh_h0[mrow][0], kt * 32, half);
      a1[kt] = gather_a_tile(&sh_h1[mrow][0], kt * 32, half);
    }
#pragma unroll 1
    for (int i = 0; i < 4; ++i) {
      int nt = wv * 4 + i, n = nt * 16 + mrow;
      f32x8 acc0, acc1;
      float bv = sh_b1[n];
#pragma unroll
      for (int r = 0; r < 8; ++r) { acc0[r] = bv; acc1[r] = 0.f; }
#pragma unroll
      for (int kt = 0; kt < 4; ++kt) {
        const bf16x16* bp =
            (const bf16x16*)(wih1p + (((size_t)kt * 32 + nt) * 32 + lane) * 16);
        acc0 = wmma_bf16(a0[kt], *bp, acc0);
        const bf16x16* bq =
            (const bf16x16*)(whh1p + (((size_t)kt * 32 + nt) * 32 + lane) * 16);
        acc1 = wmma_bf16(a1[kt], *bq, acc1);
      }
#pragma unroll
      for (int r = 0; r < 8; ++r) {
        float v = acc0[r] + acc1[r];
        v = gate_tanh ? tanhf(v) : sigm_(v);
        sh_gact[r + 8 * half][n] = v;
      }
    }
    __syncthreads();
#pragma unroll
    for (int s = 0; s < 8; ++s) {
      int j = jb + s;
      float ig = sh_gact[rr][j], fg = sh_gact[rr][kH + j];
      float gg = sh_gact[rr][2 * kH + j], og = sh_gact[rr][3 * kH + j];
      float c = fg * c1[s] + ig * gg;
      c1[s] = c;
      sh_h1[rr][j] = og * tanhf(c);
    }
    __syncthreads();

    // ---------------- logits + softmax (K=3) ----------------
    if (tid < 48) {
      int r = tid / 3, k = tid - 3 * r;
      float a = blin[k];
      for (int d = 0; d < kH; ++d) a += sh_h1[r][d] * Wlin[k * kH + d];
      sh_w[r][k] = a;
    }
    __syncthreads();
    if (tid < 16) {
      float a = sh_w[tid][0], b = sh_w[tid][1], c = sh_w[tid][2];
      float m = fmaxf(a, fmaxf(b, c));
      float ea = expf(a - m), eb = expf(b - m), ec = expf(c - m);
      float inv = 1.f / (ea + eb + ec);
      sh_w[tid][0] = ea * inv; sh_w[tid][1] = eb * inv; sh_w[tid][2] = ec * inv;
    }
    __syncthreads();

    // ---------------- emit mat_As[t+1], mat_Cs[t+1] ----------------
    {
      int r = tid >> 4;
      float w0 = sh_w[r][0], w1 = sh_w[r][1], w2 = sh_w[r][2];
      size_t rowi = (size_t)(t + 1) * kBB + bt * 16 + r;
      float* dA = outAs + rowi * 256;
      int baseA = (tid & 15) * 16;
#pragma unroll
      for (int q = 0; q < 16; ++q) {
        int ij = baseA + q;
        dA[ij] = w0 * A_K[ij] + w1 * A_K[256 + ij] + w2 * A_K[512 + ij];
      }
      float* dC = outCs + rowi * 512;
      int baseC = (tid & 15) * 32;
#pragma unroll
      for (int q = 0; q < 32; ++q) {
        int ij = baseC + q;
        dC[ij] = w0 * C_K[ij] + w1 * C_K[512 + ij] + w2 * C_K[1024 + ij];
      }
    }
    __syncthreads();
  }
}

// ---------------------------------------------------------------------------
// Kernel 4: Kalman scan.  One wave32 per batch element; fp32 WMMA (16x16x4)
// for the small matmuls, wave-cooperative Cholesky + triangular solves in LDS.
// C_t / A_{t+1} rows are pulled with the Tensor Data Mover when available.
// Streaming outputs use non-temporal stores.  All WMMA chains use two
// alternating accumulators to avoid D->C hazard NOPs.
// ---------------------------------------------------------------------------
__global__ __launch_bounds__(128) void kalman_kernel(
    const float* __restrict__ xin, const float* __restrict__ outAs,
    const float* __restrict__ outCs, const float* __restrict__ Qw,
    const float* __restrict__ Rw, const float* __restrict__ imean,
    const float* __restrict__ icov, float* __restrict__ means,
    float* __restrict__ covs, float* __restrict__ nmeans,
    float* __restrict__ ncovs) {
  __shared__ float sQ[256];
  __shared__ float sR[1024];
  __shared__ float sC[4][512];    // C_t (32x16)
  __shared__ float sA[4][256];    // A_{t+1} (16x16)
  __shared__ float sP[4][256];    // P (16x16)
  __shared__ float sS[4][1024];   // S / Cholesky L (32x32)
  __shared__ float sX[4][512];    // CP -> K^T ; [256:512) also AP
  __shared__ float sT[4][512];    // stash CP ; later [0:256)=P_f, [256:512)=KCP/PP
  __shared__ float sV[4][96];     // [0:32) innov, [32:48) m, [48:64) m_f, [64:96) a_t

  const int tid = threadIdx.x, wv = tid >> 5, lane = tid & 31;
  const int mrow = lane & 15, half = lane >> 4;
  const int b = blockIdx.x * 4 + wv;

  for (int i = tid; i < 256; i += 128) sQ[i] = Qw[i];
  for (int i = tid; i < 1024; i += 128) sR[i] = Rw[i];
  for (int i = lane; i < 256; i += 32) sP[wv][i] = icov[i];
  if (lane < 16) sV[wv][32 + lane] = imean[lane];
  __syncthreads();

  for (int t = 0; t < kTT; ++t) {
    // async DMA of this step's mixture matrices into LDS
    load_row_to_lds(&sC[wv][0], outCs + ((size_t)t * kBB + b) * 512, 512, lane);
    load_row_to_lds(&sA[wv][0], outAs + ((size_t)(t + 1) * kBB + b) * 256, 256, lane);
    sV[wv][64 + lane] = xin[((size_t)t * kBB + b) * kAD + lane];
    wait_rows();
    __syncthreads();

    // CP = C @ P  (32x16) ; stash copy in sT
#pragma unroll
    for (int mt = 0; mt < 2; ++mt) {
      f32x8 acc0, acc1;
#pragma unroll
      for (int r = 0; r < 8; ++r) { acc0[r] = 0.f; acc1[r] = 0.f; }
#pragma unroll
      for (int ks = 0; ks < 4; ++ks) {
        f32x2 a, bb;
        int k0 = ks * 4 + 2 * half;
        a.x = sC[wv][(mt * 16 + mrow) * 16 + k0];
        a.y = sC[wv][(mt * 16 + mrow) * 16 + k0 + 1];
        bb.x = sP[wv][(k0)*16 + mrow];
        bb.y = sP[wv][(k0 + 1) * 16 + mrow];
        if (ks & 1) acc1 = wmma_f32k4(a, bb, acc1);
        else        acc0 = wmma_f32k4(a, bb, acc0);
      }
#pragma unroll
      for (int r = 0; r < 8; ++r) {
        int row = mt * 16 + r + 8 * half;
        float v = acc0[r] + acc1[r];
        sX[wv][row * 16 + mrow] = v;
        sT[wv][row * 16 + mrow] = v;
      }
    }
    __syncthreads();

    // S = CP @ C^T + R  (32x32)
#pragma unroll
    for (int mt = 0; mt < 2; ++mt)
#pragma unroll
      for (int nt = 0; nt < 2; ++nt) {
        f32x8 acc0, acc1;
#pragma unroll
        for (int r = 0; r < 8; ++r) {
          acc0[r] = sR[(mt * 16 + r + 8 * half) * 32 + nt * 16 + mrow];
          acc1[r] = 0.f;
        }
#pragma unroll
        for (int ks = 0; ks < 4; ++ks) {
          f32x2 a, bb;
          int k0 = ks * 4 + 2 * half;
          a.x = sX[wv][(mt * 16 + mrow) * 16 + k0];
          a.y = sX[wv][(mt * 16 + mrow) * 16 + k0 + 1];
          bb.x = sC[wv][(nt * 16 + mrow) * 16 + k0];   // C^T[k][n] = C[n][k]
          bb.y = sC[wv][(nt * 16 + mrow) * 16 + k0 + 1];
          if (ks & 1) acc1 = wmma_f32k4(a, bb, acc1);
          else        acc0 = wmma_f32k4(a, bb, acc0);
        }
#pragma unroll
        for (int r = 0; r < 8; ++r)
          sS[wv][(mt * 16 + r + 8 * half) * 32 + nt * 16 + mrow] =
              acc0[r] + acc1[r];
      }
    __syncthreads();

    // Cholesky: S = L L^T, L lower, in place (lanes = rows)
    for (int j = 0; j < 32; ++j) {
      float d = sqrtf(fmaxf(sS[wv][j * 32 + j], 1e-18f));
      float lij = 0.f;
      if (lane >= j) {
        lij = sS[wv][lane * 32 + j] / d;
        sS[wv][lane * 32 + j] = lij;
      }
      __syncthreads();
      for (int c = j + 1; c < 32; ++c) {
        float lcj = sS[wv][c * 32 + j];
        if (lane >= c) sS[wv][lane * 32 + c] -= lij * lcj;
      }
      __syncthreads();
    }

    // Solve S X = CP  (X = K^T, 32x16), in place in sX; lanes 0..15 = RHS cols
    if (lane < 16) {
      int cc = lane;
      for (int r = 0; r < 32; ++r) {           // forward: L y = CP
        float a = sX[wv][r * 16 + cc];
        for (int p = 0; p < r; ++p) a -= sS[wv][r * 32 + p] * sX[wv][p * 16 + cc];
        sX[wv][r * 16 + cc] = a / sS[wv][r * 32 + r];
      }
      for (int r = 31; r >= 0; --r) {          // backward: L^T x = y
        float a = sX[wv][r * 16 + cc];
        for (int p = r + 1; p < 32; ++p)
          a -= sS[wv][p * 32 + r] * sX[wv][p * 16 + cc];
        sX[wv][r * 16 + cc] = a / sS[wv][r * 32 + r];
      }
    }
    __syncthreads();

    // innovation = a_t - C m ; m_f = m + K innov  (K[i][j] = X[j][i])
    {
      float al = sV[wv][64 + lane];
      float s = 0.f;
#pragma unroll
      for (int j2 = 0; j2 < 16; ++j2) s += sC[wv][lane * 16 + j2] * sV[wv][32 + j2];
      sV[wv][lane] = al - s;
    }
    __syncthreads();
    if (lane < 16) {
      float s = sV[wv][32 + lane];
      for (int j2 = 0; j2 < 32; ++j2) s += sX[wv][j2 * 16 + lane] * sV[wv][j2];
      sV[wv][48 + lane] = s;
    }
    __syncthreads();

    // KCP = (CP)^T @ X  == K C P  (symmetric) -> sT[256:512)
    {
      f32x8 acc0, acc1;
#pragma unroll
      for (int r = 0; r < 8; ++r) { acc0[r] = 0.f; acc1[r] = 0.f; }
#pragma unroll
      for (int ks = 0; ks < 8; ++ks) {
        f32x2 a, bb;
        int k0 = ks * 4 + 2 * half;
        a.x = sT[wv][(k0)*16 + mrow];       // (CP)^T[m][k] = CP[k][m]
        a.y = sT[wv][(k0 + 1) * 16 + mrow];
        bb.x = sX[wv][(k0)*16 + mrow];
        bb.y = sX[wv][(k0 + 1) * 16 + mrow];
        if (ks & 1) acc1 = wmma_f32k4(a, bb, acc1);
        else        acc0 = wmma_f32k4(a, bb, acc0);
      }
#pragma unroll
      for (int r = 0; r < 8; ++r)
        sT[wv][256 + (r + 8 * half) * 16 + mrow] = acc0[r] + acc1[r];
    }
    __syncthreads();

    // P_f = 0.5*((P - KCP) + (P - KCP)^T) -> sT[0:256)
    for (int i = lane; i < 256; i += 32) {
      int ii = i >> 4, jj = i & 15;
      float v1 = sP[wv][ii * 16 + jj] - sT[wv][256 + ii * 16 + jj];
      float v2 = sP[wv][jj * 16 + ii] - sT[wv][256 + jj * 16 + ii];
      sT[wv][i] = 0.5f * (v1 + v2);
    }
    __syncthreads();

    if (lane < 16)
      __builtin_nontemporal_store(sV[wv][48 + lane],
                                  &means[((size_t)t * kBB + b) * 16 + lane]);
    for (int i = lane; i < 256; i += 32)
      __builtin_nontemporal_store(sT[wv][i], &covs[((size_t)t * kBB + b) * 256 + i]);

    // m_p = A m_f
    if (lane < 16) {
      float s = 0.f;
#pragma unroll
      for (int j2 = 0; j2 < 16; ++j2) s += sA[wv][lane * 16 + j2] * sV[wv][48 + j2];
      sV[wv][32 + lane] = s;
      __builtin_nontemporal_store(s, &nmeans[((size_t)t * kBB + b) * 16 + lane]);
    }

    // AP = A @ P_f -> sX[256:512)
    {
      f32x8 acc0, acc1;
#pragma unroll
      for (int r = 0; r < 8; ++r) { acc0[r] = 0.f; acc1[r] = 0.f; }
#pragma unroll
      for (int ks = 0; ks < 4; ++ks) {
        f32x2 a, bb;
        int k0 = ks * 4 + 2 * half;
        a.x = sA[wv][mrow * 16 + k0];
        a.y = sA[wv][mrow * 16 + k0 + 1];
        bb.x = sT[wv][(k0)*16 + mrow];
        bb.y = sT[wv][(k0 + 1) * 16 + mrow];
        if (ks & 1) acc1 = wmma_f32k4(a, bb, acc1);
        else        acc0 = wmma_f32k4(a, bb, acc0);
      }
#pragma unroll
      for (int r = 0; r < 8; ++r)
        sX[wv][256 + (r + 8 * half) * 16 + mrow] = acc0[r] + acc1[r];
    }
    __syncthreads();

    // PP = AP @ A^T + Q -> sT[256:512), then symmetrize -> sP / next_covs
    {
      f32x8 acc0, acc1;
#pragma unroll
      for (int r = 0; r < 8; ++r) {
        acc0[r] = sQ[(r + 8 * half) * 16 + mrow];
        acc1[r] = 0.f;
      }
#pragma unroll
      for (int ks = 0; ks < 4; ++ks) {
        f32x2 a, bb;
        int k0 = ks * 4 + 2 * half;
        a.x = sX[wv][256 + mrow * 16 + k0];
        a.y = sX[wv][256 + mrow * 16 + k0 + 1];
        bb.x = sA[wv][mrow * 16 + k0];      // A^T[k][n] = A[n][k]
        bb.y = sA[wv][mrow * 16 + k0 + 1];
        if (ks & 1) acc1 = wmma_f32k4(a, bb, acc1);
        else        acc0 = wmma_f32k4(a, bb, acc0);
      }
#pragma unroll
      for (int r = 0; r < 8; ++r)
        sT[wv][256 + (r + 8 * half) * 16 + mrow] = acc0[r] + acc1[r];
    }
    __syncthreads();
    for (int i = lane; i < 256; i += 32) {
      int ii = i >> 4, jj = i & 15;
      float v = 0.5f * (sT[wv][256 + ii * 16 + jj] + sT[wv][256 + jj * 16 + ii]);
      sP[wv][i] = v;
      __builtin_nontemporal_store(v, &ncovs[((size_t)t * kBB + b) * 256 + i]);
    }
    __syncthreads();
  }
}

// ---------------------------------------------------------------------------
extern "C" void kernel_launch(void* const* d_in, const int* in_sizes, int n_in,
                              void* d_out, int out_size, void* d_ws, size_t ws_size,
                              hipStream_t stream) {
  const float* xin  = (const float*)d_in[0];
  const float* A_K  = (const float*)d_in[1];
  const float* C_K  = (const float*)d_in[2];
  const float* Q_L  = (const float*)d_in[3];
  const float* R_L  = (const float*)d_in[4];
  const float* Wih0 = (const float*)d_in[5];
  const float* Whh0 = (const float*)d_in[6];
  const float* bih0 = (const float*)d_in[7];
  const float* bhh0 = (const float*)d_in[8];
  const float* Wih1 = (const float*)d_in[9];
  const float* Whh1 = (const float*)d_in[10];
  const float* bih1 = (const float*)d_in[11];
  const float* bhh1 = (const float*)d_in[12];
  const float* Wlin = (const float*)d_in[13];
  const float* blin = (const float*)d_in[14];
  const float* imean = (const float*)d_in[15];
  const float* icov  = (const float*)d_in[16];

  char* ws = (char*)d_ws;
  __bf16* wih0p = (__bf16*)(ws + 0);        //  32 KB
  __bf16* whh0p = (__bf16*)(ws + 32768);    // 128 KB
  __bf16* wih1p = (__bf16*)(ws + 163840);   // 128 KB
  __bf16* whh1p = (__bf16*)(ws + 294912);   // 128 KB
  float* bias0 = (float*)(ws + 425984);
  float* bias1 = (float*)(ws + 428032);
  float* Qm = (float*)(ws + 430080);
  float* Rm = (float*)(ws + 431104);

  float* out = (float*)d_out;
  float* means  = out;                 // T*B*16
  float* covs   = out + 2097152;       // T*B*256
  float* nmeans = out + 35651584;      // T*B*16
  float* ncovs  = out + 37748736;      // T*B*256
  float* outAs  = out + 71303168;      // (T+1)*B*256
  float* outCs  = out + 104988672;     // (T+1)*B*512

  prep_kernel<<<842, 256, 0, stream>>>(Wih0, Whh0, bih0, bhh0, Wih1, Whh1, bih1,
                                       bhh1, Q_L, R_L, wih0p, whh0p, wih1p, whh1p,
                                       bias0, bias1, Qm, Rm);
  fill_t0_kernel<<<1536, 256, 0, stream>>>(A_K, C_K, outAs, outCs);
  lstm_fused_kernel<<<32, 256, 0, stream>>>(xin, wih0p, whh0p, wih1p, whh1p, bias0,
                                            bias1, Wlin, blin, A_K, C_K, outAs,
                                            outCs);
  kalman_kernel<<<128, 128, 0, stream>>>(xin, outAs, outCs, Qm, Rm, imean, icov,
                                         means, covs, nmeans, ncovs);
}
